// Model_37383395344917
// MI455X (gfx1250) — compile-verified, matrix operand order checked by reading, not run on hardware
//
#include <hip/hip_runtime.h>

typedef __bf16 bf16;
typedef __attribute__((ext_vector_type(16))) bf16 v16bf;
typedef __attribute__((ext_vector_type(8)))  bf16 v8bf;
typedef __attribute__((ext_vector_type(8)))  float v8f;

// ---------------------------------------------------------------------------
// WMMA helpers (gfx1250, wave32)
// ---------------------------------------------------------------------------
static __device__ __forceinline__ v8f wmma_bf16(v16bf a, v16bf b, v8f c) {
  // 8 args: (neg_a, A, neg_b, B, c_mod, C, reuse_a, reuse_b)
  return __builtin_amdgcn_wmma_f32_16x16x32_bf16(false, a, false, b, (short)0, c,
                                                 false, false);
}

// Load a 16x32 bf16 fragment (A layout; identical per-lane pattern serves the
// 32x16 B tile when the matrix is stored transposed, i.e. [N][K] row-major).
// ISA layout: lanes 0-15 hold K chunks [0..7],[16..23] of row (lane&15);
// lanes 16-31 hold K chunks [8..15],[24..31].
static __device__ __forceinline__ v16bf load_frag(const bf16* tile, int ld, int k0) {
  int lane = threadIdx.x & 31;
  int r    = lane & 15;
  int co   = (lane & 16) ? 8 : 0;
  const bf16* p = tile + (long)r * ld + k0 + co;
  v8bf lo = *(const v8bf*)(p);
  v8bf hi = *(const v8bf*)(p + 16);
  v16bf out;
#pragma unroll
  for (int i = 0; i < 8; ++i) { out[i] = lo[i]; out[i + 8] = hi[i]; }
  return out;
}

// ---------------------------------------------------------------------------
// Generic WMMA GEMM:  C[M,N] (f32) = act( A[M,K](bf16) * Bt[N,K]^T (bf16) + bias )
// Block tile 128x64, 8 waves (4x2), each wave 32x32 (2x2 WMMA tiles).
// M multiple of 128, N multiple of 64, K multiple of 32. grid.z = batch.
// ---------------------------------------------------------------------------
__global__ __launch_bounds__(256) void gemm_bt(
    const bf16* __restrict__ A, const bf16* __restrict__ Bt,
    const float* __restrict__ bias, float* __restrict__ C,
    int K, int lda, int ldb, int ldc,
    long long sA, long long sB, long long sC, int act)
{
  long long z = blockIdx.z;
  A += z * sA; Bt += z * sB; C += z * sC;
  int wave = threadIdx.x >> 5;
  int wm = wave & 3, wn = wave >> 2;
  int m0 = blockIdx.x * 128 + wm * 32;
  int n0 = blockIdx.y * 64  + wn * 32;

  v8f acc00 = {0,0,0,0,0,0,0,0};
  v8f acc01 = acc00, acc10 = acc00, acc11 = acc00;

  for (int k0 = 0; k0 < K; k0 += 32) {
    v16bf a0 = load_frag(A + (long)m0        * lda, lda, k0);
    v16bf a1 = load_frag(A + (long)(m0 + 16) * lda, lda, k0);
    v16bf b0 = load_frag(Bt + (long)n0        * ldb, ldb, k0);
    v16bf b1 = load_frag(Bt + (long)(n0 + 16) * ldb, ldb, k0);
    acc00 = wmma_bf16(a0, b0, acc00);
    acc01 = wmma_bf16(a0, b1, acc01);
    acc10 = wmma_bf16(a1, b0, acc10);
    acc11 = wmma_bf16(a1, b1, acc11);
  }

  int lane = threadIdx.x & 31;
  int col  = lane & 15;
  int hi8  = (lane & 16) ? 8 : 0;
  v8f accs[2][2] = {{acc00, acc01}, {acc10, acc11}};
#pragma unroll
  for (int i = 0; i < 2; ++i)
#pragma unroll
    for (int j = 0; j < 2; ++j) {
      int n = n0 + j * 16 + col;
      float bv = bias ? bias[n] : 0.f;
#pragma unroll
      for (int r = 0; r < 8; ++r) {
        float v = accs[i][j][r] + bv;
        if (act) v = tanhf(v);
        C[(long)(m0 + i * 16 + hi8 + r) * ldc + n] = v;
      }
    }
}

// ---------------------------------------------------------------------------
// LSTM scan: one workgroup per (question,direction). 16 waves; each wave owns
// 5 of the 80 (m,n) 16x16 tiles of the [128 x 4*160] gate matrix and keeps the
// cell state c for those elements in VGPRs across all 128 steps.
// Whh (padded bf16 [640 x 160], gate-blocked) streams from L2; h (bf16) lives
// in LDS; Zpre (with biases folded in) is the WMMA C operand of k-step 0.
// ---------------------------------------------------------------------------
__global__ __launch_bounds__(512) void lstm_kernel(
    const float* __restrict__ zpre,  // [4][16384][640]
    const bf16*  __restrict__ whhp,  // [2][640][160]
    const int* __restrict__ l1, const int* __restrict__ l2,
    float* __restrict__ r1, float* __restrict__ r2)   // [16384][320]
{
  __shared__ __align__(16) bf16 hbuf[128 * 160];

  int seq  = blockIdx.x;      // 0:q1 fwd 1:q1 bwd 2:q2 fwd 3:q2 bwd
  int dirb = seq & 1;
  const float* zp  = zpre + (long)seq * 16384 * 640;
  const bf16*  W   = whhp + (long)dirb * 640 * 160;
  float*       rout = (seq >> 1) ? r2 : r1;
  const int*   lens = (seq >> 1) ? l2 : l1;

  for (int i = threadIdx.x; i < 128 * 160; i += 512) hbuf[i] = (bf16)0.f;
  __syncthreads();

  int wave = threadIdx.x >> 5;
  int lane = threadIdx.x & 31;
  int col  = lane & 15;
  int hi8  = (lane & 16) ? 8 : 0;

  int pm[5], pn[5];
  v8f cst[5];
#pragma unroll
  for (int p = 0; p < 5; ++p) {
    int idx = wave * 5 + p;
    pm[p] = idx / 10; pn[p] = idx % 10;
    cst[p] = (v8f){0,0,0,0,0,0,0,0};
  }

  for (int t = 0; t < 128; ++t) {
    v8f hnew[5];
#pragma unroll
    for (int p = 0; p < 5; ++p) {
      int m0 = pm[p] * 16, n0 = pn[p] * 16;
      v16bf afr[5];
#pragma unroll
      for (int kt = 0; kt < 5; ++kt)
        afr[kt] = load_frag(hbuf + m0 * 160, 160, kt * 32);

      v8f zg[4];
#pragma unroll
      for (int g = 0; g < 4; ++g) {
        int gc = g * 160 + n0;
        v8f acc;
#pragma unroll
        for (int r = 0; r < 8; ++r) {
          int b = m0 + hi8 + r;
          acc[r] = zp[(((long)b << 7) + t) * 640 + gc + col];
        }
#pragma unroll
        for (int kt = 0; kt < 5; ++kt) {
          v16bf bfr = load_frag(W + (long)gc * 160, 160, kt * 32);
          acc = wmma_bf16(afr[kt], bfr, acc);
        }
        zg[g] = acc;
      }

      v8f hn;
#pragma unroll
      for (int r = 0; r < 8; ++r) {
        float ig = 1.f / (1.f + __expf(-zg[0][r]));
        float fg = 1.f / (1.f + __expf(-zg[1][r]));
        float gg = tanhf(zg[2][r]);
        float og = 1.f / (1.f + __expf(-zg[3][r]));
        float c  = fg * cst[p][r] + ig * gg;
        cst[p][r] = c;
        hn[r] = og * tanhf(c);
      }
      hnew[p] = hn;

      int j = n0 + col;
      if (j < 150) {
#pragma unroll
        for (int r = 0; r < 8; ++r) {
          int b = m0 + hi8 + r;
          int len = lens[b];
          int tt = dirb ? ((t < len) ? (len - 1 - t) : t) : t;   // un-reverse bwd
          rout[(((long)b << 7) + tt) * 320 + dirb * 150 + j] = hn[r];
        }
      }
    }
    __syncthreads();               // all reads of old h done
#pragma unroll
    for (int p = 0; p < 5; ++p) {
      int j = pn[p] * 16 + col;
#pragma unroll
      for (int r = 0; r < 8; ++r) {
        int b = pm[p] * 16 + hi8 + r;
        hbuf[(long)b * 160 + j] = (bf16)hnew[p][r];
      }
    }
    __syncthreads();               // new h visible for next step
  }
}

// ---------------------------------------------------------------------------
// Support kernels
// ---------------------------------------------------------------------------
__global__ void pad_gate(const float* __restrict__ W, bf16* __restrict__ out,
                         int Kin, int Kpad)
{
  long idx = (long)blockIdx.x * 256 + threadIdx.x;
  if (idx >= 640L * Kpad) return;
  int rp = (int)(idx / Kpad), k = (int)(idx % Kpad);
  int g = rp / 160, j = rp % 160;
  float v = (j < 150 && k < Kin) ? W[(long)(g * 150 + j) * Kin + k] : 0.f;
  out[idx] = (bf16)v;
}

__global__ void gate_bias(const float* bih, const float* bhh, float* out)
{
  int rp = blockIdx.x * 256 + threadIdx.x;
  if (rp >= 640) return;
  int g = rp / 160, j = rp % 160;
  out[rp] = (j < 150) ? (bih[g * 150 + j] + bhh[g * 150 + j]) : 0.f;
}

__global__ void pad_fc(const float* __restrict__ W, const float* __restrict__ bsrc,
                       bf16* __restrict__ out, float* __restrict__ bout)
{
  long idx = (long)blockIdx.x * 256 + threadIdx.x;
  if (idx >= 320L * 608) return;
  int r = (int)(idx / 608), c = (int)(idx % 608);
  out[idx] = (bf16)((r < 300 && c < 600) ? W[(long)r * 600 + c] : 0.f);
  if (c == 0) bout[r] = (r < 300) ? bsrc[r] : 0.f;
}

__global__ void embed_kernel(const int* q1, const int* q2, const int* l1,
                             const int* l2, const float* __restrict__ emb,
                             bf16* __restrict__ xbf)
{
  int row = blockIdx.x, seq = blockIdx.y;
  int b = row >> 7, t = row & 127;
  const int* q = (seq >> 1) ? q2 : q1;
  const int* L = (seq >> 1) ? l2 : l1;
  int tt = t;
  if (seq & 1) { int len = L[b]; tt = (t < len) ? (len - 1 - t) : t; }
  long tok = q[b * 128 + tt];
  bf16* dst = xbf + ((long)seq * 16384 + row) * 320;
  for (int e = threadIdx.x; e < 320; e += 128)
    dst[e] = (e < 300) ? (bf16)emb[tok * 300 + e] : (bf16)0.f;
}

__global__ void s12_kernel(const float* r1, const float* r2, const float* attw,
                           float* s1, float* s2)
{
  int sel = blockIdx.y;
  int row = blockIdx.x * 128 + threadIdx.x;
  const float* r = sel ? r2 : r1;
  const float* w = attw + sel * 300;
  float acc = 0.f;
  for (int d = 0; d < 300; ++d) acc += r[(long)row * 320 + d] * w[d];
  (sel ? s2 : s1)[row] = acc;
}

__global__ void transpose_r(const float* r1, const float* r2,
                            bf16* r1t, bf16* r2t)
{
  int b = blockIdx.x, sel = blockIdx.y;
  const float* r = sel ? r2 : r1;
  bf16* rt = sel ? r2t : r1t;
  for (int idx = threadIdx.x; idx < 320 * 128; idx += 256) {
    int d = idx >> 7, k = idx & 127;
    float v = (d < 300) ? r[((long)b * 128 + k) * 320 + d] : 0.f;
    rt[((long)b * 320 + d) * 128 + k] = (bf16)v;
  }
}

__global__ void attn_softmax(const float* s1, const float* s2, const float* attb,
                             const int* l1, const int* l2, bf16* p1, bf16* p2)
{
  __shared__ float sm[128];
  int qi = blockIdx.x, b = blockIdx.y, sel = blockIdx.z, k = threadIdx.x;
  float ab = attb[0];
  int lenk = sel ? l1[b] : l2[b];
  float sq = sel ? s2[b * 128 + qi] : s1[b * 128 + qi];
  float sk = sel ? s1[b * 128 + k]  : s2[b * 128 + k];
  float logit = (k < lenk) ? tanhf(sq + sk + ab) : -1e9f;
  sm[k] = logit; __syncthreads();
  for (int s = 64; s > 0; s >>= 1) { if (k < s) sm[k] = fmaxf(sm[k], sm[k + s]); __syncthreads(); }
  float mx = sm[0]; __syncthreads();
  float e = __expf(logit - mx);
  sm[k] = e; __syncthreads();
  for (int s = 64; s > 0; s >>= 1) { if (k < s) sm[k] += sm[k + s]; __syncthreads(); }
  float inv = 1.f / sm[0];
  (sel ? p2 : p1)[((long)b * 128 + qi) * 128 + k] = (bf16)(e * inv);
}

__global__ void comp_build(const float* __restrict__ r, const float* __restrict__ ar,
                           bf16* __restrict__ comp)
{
  long row = blockIdx.x;
  for (int c = threadIdx.x; c < 608; c += 256) {
    float v = 0.f;
    if (c < 300)      { float a = r[row * 320 + c];       v = fabsf(a - ar[row * 320 + c]); }
    else if (c < 600) { int d = c - 300;                  v = ar[row * 320 + d] * r[row * 320 + d]; }
    comp[row * 608 + c] = (bf16)v;
  }
}

__global__ void zero_bf16(bf16* p, long n)
{
  long i = (long)blockIdx.x * 256 + threadIdx.x;
  if (i < n) p[i] = (bf16)0.f;
}

__global__ void mean_kernel(const float* __restrict__ compfc, const int* lens,
                            bf16* __restrict__ hj, int colbase)
{
  int b = blockIdx.x, d = threadIdx.x;
  if (d >= 300) return;
  int len = lens[b];
  float s = 0.f;
  for (int t = 0; t < len; ++t) s += compfc[((long)b * 128 + t) * 320 + d];
  hj[(long)b * 608 + colbase + d] = (bf16)(s / (float)len);
}

__global__ void final_fc(const float* __restrict__ joint, const float* w,
                         const float* b, float* out)
{
  int bb = blockIdx.x, o = threadIdx.x;
  if (o < 2) {
    float s = b[o];
    for (int d = 0; d < 300; ++d) s += joint[(long)bb * 320 + d] * w[o * 300 + d];
    out[bb * 2 + o] = s;
  }
}

// ---------------------------------------------------------------------------
static inline int cdiv_i(long a, long b) { return (int)((a + b - 1) / b); }

extern "C" void kernel_launch(void* const* d_in, const int* in_sizes, int n_in,
                              void* d_out, int out_size, void* d_ws, size_t ws_size,
                              hipStream_t stream)
{
  (void)in_sizes; (void)n_in; (void)out_size; (void)ws_size;
  const int*   q1    = (const int*)  d_in[0];
  const int*   q2    = (const int*)  d_in[1];
  const int*   l1    = (const int*)  d_in[2];
  const int*   l2    = (const int*)  d_in[3];
  const float* emb   = (const float*)d_in[4];
  const float* Wih_f = (const float*)d_in[5];
  const float* Whh_f = (const float*)d_in[6];
  const float* bih_f = (const float*)d_in[7];
  const float* bhh_f = (const float*)d_in[8];
  const float* Wih_b = (const float*)d_in[9];
  const float* Whh_b = (const float*)d_in[10];
  const float* bih_b = (const float*)d_in[11];
  const float* bhh_b = (const float*)d_in[12];
  const float* att_w = (const float*)d_in[13];
  const float* att_b = (const float*)d_in[14];
  const float* fc0_w = (const float*)d_in[15];
  const float* fc0_b = (const float*)d_in[16];
  const float* fc1_w = (const float*)d_in[17];
  const float* fc1_b = (const float*)d_in[18];
  const float* fc2_w = (const float*)d_in[19];
  const float* fc2_b = (const float*)d_in[20];
  float* out = (float*)d_out;
  char* ws = (char*)d_ws;

  // ---- workspace layout (bytes) ----
  const size_t SZ_XBF  = 4ULL * 16384 * 320 * 2;   //  41.9 MB
  const size_t SZ_ZPRE = 4ULL * 16384 * 640 * 4;   // 167.8 MB (reused below)
  const size_t SZ_R    = 16384ULL * 320 * 4;       //  21.0 MB each
  size_t o_xbf  = 0;
  size_t o_zpre = o_xbf + SZ_XBF;
  // aliased into the dead Zpre region after the LSTM completes:
  size_t o_ar1     = o_zpre;
  size_t o_ar2     = o_ar1     + 16384ULL * 320 * 4;
  size_t o_compfc1 = o_ar2     + 16384ULL * 320 * 4;
  size_t o_compfc2 = o_compfc1 + 16384ULL * 320 * 4;
  size_t o_comp1   = o_compfc2 + 16384ULL * 320 * 4;
  size_t o_comp2   = o_comp1   + 16384ULL * 608 * 2;
  size_t o_r1t     = o_comp2   + 16384ULL * 608 * 2;
  size_t o_r2t     = o_r1t     + 128ULL * 320 * 128 * 2;
  size_t o_p1      = o_r2t     + 128ULL * 320 * 128 * 2;
  size_t o_p2      = o_p1      + 128ULL * 128 * 128 * 2;
  size_t o_r1   = o_zpre + SZ_ZPRE;
  size_t o_r2   = o_r1  + SZ_R;
  size_t o_wih  = o_r2  + SZ_R;                 // [2][640*320] bf16
  size_t o_whh  = o_wih  + 2ULL * 640 * 320 * 2; // [2][640*160] bf16
  size_t o_bias = o_whh  + 2ULL * 640 * 160 * 2; // [2][640] f32
  size_t o_fc0w = o_bias + 2ULL * 640 * 4;       // [320*608] bf16
  size_t o_fc0b = o_fc0w + 320ULL * 608 * 2;     // [320] f32
  size_t o_fc1w = o_fc0b + 320ULL * 4;
  size_t o_fc1b = o_fc1w + 320ULL * 608 * 2;
  size_t o_s1   = o_fc1b + 320ULL * 4;           // [16384] f32
  size_t o_s2   = o_s1   + 16384ULL * 4;
  size_t o_hj   = o_s2   + 16384ULL * 4;         // [128*608] bf16
  size_t o_jnt  = o_hj   + 128ULL * 608 * 2;     // [128*320] f32

  bf16*  xbf   = (bf16*) (ws + o_xbf);
  float* zpre  = (float*)(ws + o_zpre);
  float* ar1   = (float*)(ws + o_ar1);
  float* ar2   = (float*)(ws + o_ar2);
  float* cfc1  = (float*)(ws + o_compfc1);
  float* cfc2  = (float*)(ws + o_compfc2);
  bf16*  comp1 = (bf16*) (ws + o_comp1);
  bf16*  comp2 = (bf16*) (ws + o_comp2);
  bf16*  r1t   = (bf16*) (ws + o_r1t);
  bf16*  r2t   = (bf16*) (ws + o_r2t);
  bf16*  p1    = (bf16*) (ws + o_p1);
  bf16*  p2    = (bf16*) (ws + o_p2);
  float* r1    = (float*)(ws + o_r1);
  float* r2    = (float*)(ws + o_r2);
  bf16*  wihp  = (bf16*) (ws + o_wih);
  bf16*  whhp  = (bf16*) (ws + o_whh);
  float* biasp = (float*)(ws + o_bias);
  bf16*  fc0wp = (bf16*) (ws + o_fc0w);
  float* fc0bp = (float*)(ws + o_fc0b);
  bf16*  fc1wp = (bf16*) (ws + o_fc1w);
  float* fc1bp = (float*)(ws + o_fc1b);
  float* s1    = (float*)(ws + o_s1);
  float* s2    = (float*)(ws + o_s2);
  bf16*  hj    = (bf16*) (ws + o_hj);
  float* joint = (float*)(ws + o_jnt);

  // 1) weight prep (bf16, gate-blocked padding)
  pad_gate<<<cdiv_i(640L * 320, 256), 256, 0, stream>>>(Wih_f, wihp, 300, 320);
  pad_gate<<<cdiv_i(640L * 320, 256), 256, 0, stream>>>(Wih_b, wihp + 640L * 320, 300, 320);
  pad_gate<<<cdiv_i(640L * 160, 256), 256, 0, stream>>>(Whh_f, whhp, 150, 160);
  pad_gate<<<cdiv_i(640L * 160, 256), 256, 0, stream>>>(Whh_b, whhp + 640L * 160, 150, 160);
  gate_bias<<<3, 256, 0, stream>>>(bih_f, bhh_f, biasp);
  gate_bias<<<3, 256, 0, stream>>>(bih_b, bhh_b, biasp + 640);
  pad_fc<<<cdiv_i(320L * 608, 256), 256, 0, stream>>>(fc0_w, fc0_b, fc0wp, fc0bp);
  pad_fc<<<cdiv_i(320L * 608, 256), 256, 0, stream>>>(fc1_w, fc1_b, fc1wp, fc1bp);

  // 2) embedding gather (fwd + reversed) -> bf16 [4][16384][320]
  embed_kernel<<<dim3(16384, 4), 128, 0, stream>>>(q1, q2, l1, l2, emb, xbf);

  // 3) input projections Zpre = X @ Wih^T + (bih+bhh)   [16384 x 640] x4
  for (int s = 0; s < 4; ++s)
    gemm_bt<<<dim3(128, 10, 1), 256, 0, stream>>>(
        xbf + (long)s * 16384 * 320, wihp + (long)(s & 1) * 640 * 320,
        biasp + (s & 1) * 640, zpre + (long)s * 16384 * 640,
        320, 320, 320, 640, 0, 0, 0, 0);

  // 4) BiLSTM scan (1 WGP per direction, WMMA recurrent GEMM)
  lstm_kernel<<<4, 512, 0, stream>>>(zpre, whhp, l1, l2, r1, r2);

  // 5) attention scores + per-batch transposed bf16 copies of r
  s12_kernel<<<dim3(128, 2), 128, 0, stream>>>(r1, r2, att_w, s1, s2);
  transpose_r<<<dim3(128, 2), 256, 0, stream>>>(r1, r2, r1t, r2t);

  // 6) masked softmax over keys (both directions), bf16 probs
  attn_softmax<<<dim3(128, 128, 2), 128, 0, stream>>>(s1, s2, att_b, l1, l2, p1, p2);

  // 7) attended vectors: ar = probs @ r_k  (batched WMMA GEMM, K=128)
  gemm_bt<<<dim3(1, 5, 128), 256, 0, stream>>>(p1, r2t, nullptr, ar1,
      128, 128, 128, 320, 128LL * 128, 320LL * 128, 128LL * 320, 0);
  gemm_bt<<<dim3(1, 5, 128), 256, 0, stream>>>(p2, r1t, nullptr, ar2,
      128, 128, 128, 320, 128LL * 128, 320LL * 128, 128LL * 320, 0);

  // 8) comp = [|r - ar|, ar * r] (bf16, padded to 608)
  comp_build<<<16384, 256, 0, stream>>>(r1, ar1, comp1);
  comp_build<<<16384, 256, 0, stream>>>(r2, ar2, comp2);

  // 9) fc0 + tanh (WMMA GEMM, K=608)
  gemm_bt<<<dim3(128, 5, 1), 256, 0, stream>>>(comp1, fc0wp, fc0bp, cfc1,
      608, 608, 608, 320, 0, 0, 0, 1);
  gemm_bt<<<dim3(128, 5, 1), 256, 0, stream>>>(comp2, fc0wp, fc0bp, cfc2,
      608, 608, 608, 320, 0, 0, 0, 1);

  // 10) masked mean over time -> hjoint bf16 [128 x 608]
  zero_bf16<<<cdiv_i(128L * 608, 256), 256, 0, stream>>>(hj, 128L * 608);
  mean_kernel<<<128, 320, 0, stream>>>(cfc1, l1, hj, 0);
  mean_kernel<<<128, 320, 0, stream>>>(cfc2, l2, hj, 300);

  // 11) fc1 + tanh (WMMA GEMM), then fc2
  gemm_bt<<<dim3(1, 5, 1), 256, 0, stream>>>(hj, fc1wp, fc1bp, joint,
      608, 608, 608, 320, 0, 0, 0, 1);
  final_fc<<<128, 32, 0, stream>>>(joint, fc2_w, fc2_b, out);
}
